// Layer3Model_11373073400010
// MI455X (gfx1250) — matrix operand order repaired and checked by reading, not compiled, not run on hardware
//
#include <hip/hip_runtime.h>
#include <cstdint>
#include <cstddef>

// ---------------- model constants ----------------
#define BATCH   16384
#define SEQT    11
#define NTOK    (BATCH * SEQT)     // 180224 = 1408 * 128
#define DIM     128
#define FFDIM   512
#define NQKV    384                // Q(128) | K(128) | V(128)
#define TOUTDIM 120                // T_OUT * STATE_DIM
#define SCALE_ATTN 0.17677669529663687f   // 1/sqrt(32)

typedef __attribute__((ext_vector_type(16))) _Float16     v16h;
typedef __attribute__((ext_vector_type(8)))  float        v8f;
typedef __attribute__((ext_vector_type(4)))  unsigned int v4u;

union FragU { v16h h; v4u u[2]; };

#define LDAH 136   // LDS row stride in halves (128 + 8 pad, 16B aligned rows)
#define GEMM_LDS_BYTES (128 * LDAH * 2)          // 34816
#define FFN_LDS_BYTES  (2 * 128 * LDAH * 2)      // 69632

// ---------------- CDNA5 async global->LDS staging ----------------
// GLOBAL_LOAD_ASYNC_TO_LDS_B128: per-lane global address -> per-lane LDS address,
// no VGPR round trip; tracked with ASYNCcnt (invisible to the compiler, so we
// drain explicitly with s_wait_asynccnt before the workgroup barrier).
__device__ __forceinline__ void asyncLdsCopyB128(void* ldsDst, const void* gsrc) {
    unsigned l = (unsigned)(uintptr_t)ldsDst;   // generic shared ptr: low 32 bits = LDS byte addr
    asm volatile("global_load_async_to_lds_b128 %0, %1, off"
                 :: "v"(l), "v"(gsrc)
                 : "memory");
}
__device__ __forceinline__ void asyncWaitAll() {
    asm volatile("s_wait_asynccnt 0x0" ::: "memory");
}

// ---------------- helpers ----------------
__device__ __forceinline__ float waveRed(float v) {
#pragma unroll
    for (int m = 16; m >= 1; m >>= 1) v += __shfl_xor(v, m, 32);
    return v;
}

// LayerNorm (pre) + Minkowski quadratic form for one token; wave = 32 lanes x 4 dims
__device__ __forceinline__ void mln_core(const float* __restrict__ xrow,
                                         const float* __restrict__ pg,
                                         const float* __restrict__ pb,
                                         int lane, float4* yOut, float* mqOut) {
    float4 v = *(const float4*)(xrow + lane * 4);
    float mu = waveRed(v.x + v.y + v.z + v.w) * (1.0f / 128.0f);
    float dx0 = v.x - mu, dx1 = v.y - mu, dx2 = v.z - mu, dx3 = v.w - mu;
    float var = waveRed(dx0 * dx0 + dx1 * dx1 + dx2 * dx2 + dx3 * dx3) * (1.0f / 128.0f);
    float inv = rsqrtf(var + 1e-5f);
    float4 pgv = *(const float4*)(pg + lane * 4);
    float4 pbv = *(const float4*)(pb + lane * 4);
    float4 y;
    y.x = pgv.x * dx0 * inv + pbv.x;
    y.y = pgv.y * dx1 * inv + pbv.y;
    y.z = pgv.z * dx2 * inv + pbv.z;
    y.w = pgv.w * dx3 * inv + pbv.w;
    float q = y.x * y.x + y.y * y.y + y.z * y.z + y.w * y.w;
    // dims 0..63 = time part (negative), 64..127 = space part (positive)
    float mq = waveRed(lane < 16 ? -q : q);
    *yOut = y;
    *mqOut = mq;
}

// ---------------- kernels ----------------

// pack fp32 W[K][N] -> f16 Wt[n][k] (rows n >= N zero-padded)
__global__ __launch_bounds__(256) void packWT(const float* __restrict__ W,
                                              _Float16* __restrict__ out,
                                              int K, int N, int Nrows) {
    int i = blockIdx.x * 256 + threadIdx.x;
    if (i >= Nrows * K) return;
    int n = i / K, k = i - n * K;
    out[i] = (n < N) ? (_Float16)W[(size_t)k * N + n] : (_Float16)0.0f;
}

// h = x @ We + be + pos[t]
__global__ __launch_bounds__(256) void embed_kernel(const float* __restrict__ x,
                                                    const float* __restrict__ We,
                                                    const float* __restrict__ be,
                                                    const float* __restrict__ pos,
                                                    float* __restrict__ h) {
    size_t idx = (size_t)blockIdx.x * 256 + threadIdx.x;
    if (idx >= (size_t)NTOK * DIM) return;
    int tok = (int)(idx >> 7), d = (int)(idx & 127);
    int t = tok % SEQT;
    float acc = be[d] + pos[t * DIM + d];
    const float* xr = x + (size_t)tok * 12;
#pragma unroll
    for (int k = 0; k < 12; ++k) acc += xr[k] * We[k * DIM + d];
    h[idx] = acc;
}

// pass 1: per-block partial sums of |mq|
__global__ __launch_bounds__(256) void mln_pass1(const float* __restrict__ h,
                                                 const float* __restrict__ pg,
                                                 const float* __restrict__ pb,
                                                 float* __restrict__ partial) {
    __shared__ float wsum[8];
    int wave = threadIdx.x >> 5, lane = threadIdx.x & 31;
    size_t tok = (size_t)blockIdx.x * 8 + wave;
    float4 y; float mq;
    mln_core(h + tok * DIM, pg, pb, lane, &y, &mq);
    if (lane == 0) wsum[wave] = fabsf(mq);
    __syncthreads();
    if (threadIdx.x == 0) {
        float t = 0.f;
#pragma unroll
        for (int i = 0; i < 8; ++i) t += wsum[i];
        partial[blockIdx.x] = t;
    }
}

// deterministic fixed-order reduction -> eps = max(0.01*mean|mq|, 1e-5)
__global__ __launch_bounds__(256) void reduce_eps(const float* __restrict__ partial,
                                                  int n, float* __restrict__ epsOut) {
    __shared__ float sm[256];
    float s = 0.f;
    for (int i = threadIdx.x; i < n; i += 256) s += partial[i];
    sm[threadIdx.x] = s;
    __syncthreads();
    for (int off = 128; off > 0; off >>= 1) {
        if (threadIdx.x < off) sm[threadIdx.x] += sm[threadIdx.x + off];
        __syncthreads();
    }
    if (threadIdx.x == 0) {
        float mean = sm[0] * (1.0f / (float)NTOK);
        epsOut[0] = fmaxf(0.01f * mean, 1e-5f);
    }
}

// pass 2: apply Minkowski LN, emit f16 activations (optionally only last token rows)
__global__ __launch_bounds__(256) void mln_pass2(const float* __restrict__ h,
                                                 const float* __restrict__ pg,
                                                 const float* __restrict__ pb,
                                                 const float* __restrict__ g,
                                                 const float* __restrict__ b,
                                                 const float* __restrict__ epsPtr,
                                                 _Float16* __restrict__ out,
                                                 int lastOnly) {
    int wave = threadIdx.x >> 5, lane = threadIdx.x & 31;
    size_t tok = (size_t)blockIdx.x * 8 + wave;
    float4 y; float mq;
    mln_core(h + tok * DIM, pg, pb, lane, &y, &mq);
    float eps = epsPtr[0];
    float inv = 1.0f / (sqrtf(fabsf(mq) + eps) + eps);
    float4 gv = *(const float4*)(g + lane * 4);
    float4 bv = *(const float4*)(b + lane * 4);
    size_t row = tok;
    if (lastOnly) {
        if ((tok % SEQT) != (SEQT - 1)) return;
        row = tok / SEQT;
    }
    _Float16* p = out + row * DIM + lane * 4;
    p[0] = (_Float16)(gv.x * y.x * inv + bv.x);
    p[1] = (_Float16)(gv.y * y.y * inv + bv.y);
    p[2] = (_Float16)(gv.z * y.z * inv + bv.z);
    p[3] = (_Float16)(gv.w * y.w * inv + bv.w);
}

// Generic K=128 WMMA GEMM: C[M][N] = A[M][128] @ Wt^T, 128-row tile / block, 8 waves.
// MODE 0: store f16 to out (ldOut = N)        [QKV projection]
// MODE 1: resid[row*128+col] += acc + bias    [attention out-proj + residual]
// MODE 2: out f32 [row*storeN+col] = acc+bias for col < storeN   [traj head]
template <int MODE>
__global__ __launch_bounds__(256) void gemm128(const _Float16* __restrict__ A,
                                               const _Float16* __restrict__ Wt,
                                               const float* __restrict__ bias,
                                               void* __restrict__ outp,
                                               float* __restrict__ resid,
                                               int N, int ldOut, int storeN) {
    extern __shared__ _Float16 lds[];
    int tid = threadIdx.x;
    size_t rowBase = (size_t)blockIdx.x * 128;
    const _Float16* Ag = A + rowBase * DIM;
    for (int i = tid; i < 2048; i += 256) {   // 128 rows x 16 x (8 halves), async DMA to LDS
        int r = i >> 4, c = i & 15;
        asyncLdsCopyB128(&lds[r * LDAH + c * 8], Ag + (size_t)r * DIM + c * 8);
    }
    asyncWaitAll();
    __syncthreads();
    int wave = tid >> 5, lane = tid & 31;
    int laneLo = lane & 15, laneHi = lane >> 4;
    int wr = wave << 4;
    for (int nt = 0; nt < (N >> 4); ++nt) {
        v8f acc = {};
#pragma unroll
        for (int kt = 0; kt < 4; ++kt) {
            int kb = kt << 5;
            const _Float16* pa = &lds[(wr + laneLo) * LDAH + kb + laneHi * 8];
            FragU fa; fa.u[0] = *(const v4u*)pa; fa.u[1] = *(const v4u*)(pa + 16);
            const _Float16* pb = &Wt[(size_t)((nt << 4) + laneLo) * DIM + kb + laneHi * 8];
            FragU fb; fb.u[0] = *(const v4u*)pb; fb.u[1] = *(const v4u*)(pb + 16);
            acc = __builtin_amdgcn_wmma_f32_16x16x32_f16(false, fa.h, false, fb.h,
                                                         (short)0, acc, false, false);
        }
        int col = (nt << 4) + laneLo;
#pragma unroll
        for (int j = 0; j < 8; ++j) {
            size_t row = rowBase + wr + laneHi * 8 + j;
            float v = acc[j];
            if (MODE == 0) {
                ((_Float16*)outp)[row * (size_t)ldOut + col] = (_Float16)v;
            } else if (MODE == 1) {
                float* p = resid + row * (size_t)DIM + col;
                *p = *p + v + bias[col];
            } else {
                if (col < storeN)
                    ((float*)outp)[row * (size_t)storeN + col] = v + bias[col];
            }
        }
    }
}

// per-(batch,head) attention over T=11; one wave per task, K/V staged in LDS
__global__ __launch_bounds__(256) void attn_kernel(const _Float16* __restrict__ qkv,
                                                   _Float16* __restrict__ ctx,
                                                   const float* __restrict__ wsig) {
    __shared__ float kv[8][2][SEQT * 32];
    int wave = threadIdx.x >> 5, lane = threadIdx.x & 31;
    int task = blockIdx.x * 8 + wave;
    int b = task >> 2, hh = task & 3;
    size_t base = (size_t)b * SEQT * NQKV;
    float* Ks = kv[wave][0];
    float* Vs = kv[wave][1];
    for (int i = lane; i < SEQT * 32; i += 32) {
        int s = i >> 5, d = i & 31;
        Ks[i] = (float)qkv[base + (size_t)s * NQKV + 128 + hh * 32 + d];
        Vs[i] = (float)qkv[base + (size_t)s * NQKV + 256 + hh * 32 + d];
    }
    __syncthreads();
    float sig = 1.0f / (1.0f + expf(-wsig[0]));
    int t = lane;
    if (t < SEQT) {
        float q[32];
#pragma unroll
        for (int d = 0; d < 32; ++d) q[d] = (float)qkv[base + (size_t)t * NQKV + hh * 32 + d];
        float sc[SEQT];
        float mx = -3.0e38f;
#pragma unroll
        for (int s = 0; s < SEQT; ++s) {
            float dot = 0.f;
#pragma unroll
            for (int d = 0; d < 32; ++d) dot += q[d] * Ks[s * 32 + d];
            float v = (hh < 2) ? (-sig * dot) : dot;
            v *= SCALE_ATTN;
            sc[s] = v;
            mx = fmaxf(mx, v);
        }
        float sum = 0.f;
#pragma unroll
        for (int s = 0; s < SEQT; ++s) { float e = expf(sc[s] - mx); sc[s] = e; sum += e; }
        float inv = 1.0f / sum;
        _Float16* orow = ctx + ((size_t)b * SEQT + t) * DIM + hh * 32;
#pragma unroll
        for (int d = 0; d < 32; ++d) {
            float o = 0.f;
#pragma unroll
            for (int s = 0; s < SEQT; ++s) o += sc[s] * Vs[s * 32 + d];
            orow[d] = (_Float16)(o * inv);
        }
    }
}

// Fused FFN: h += gelu(ln @ W1 + b1) @ W2 + b2. 128-token tile, GELU intermediate
// kept in LDS in 128-wide K-chunks; ff2 accumulators held across chunks in VGPRs.
__global__ __launch_bounds__(256) void ffn_kernel(const _Float16* __restrict__ A,
                                                  const _Float16* __restrict__ W1t,
                                                  const float* __restrict__ b1,
                                                  const _Float16* __restrict__ W2t,
                                                  const float* __restrict__ b2,
                                                  float* __restrict__ h) {
    extern __shared__ _Float16 lds[];
    _Float16* ldsA = lds;                 // 128 x LDAH
    _Float16* ldsH = lds + 128 * LDAH;    // 128 x LDAH (gelu chunk)
    int tid = threadIdx.x;
    size_t rowBase = (size_t)blockIdx.x * 128;
    const _Float16* Ag = A + rowBase * DIM;
    for (int i = tid; i < 2048; i += 256) {   // async DMA stage of A tile
        int r = i >> 4, c = i & 15;
        asyncLdsCopyB128(&ldsA[r * LDAH + c * 8], Ag + (size_t)r * DIM + c * 8);
    }
    asyncWaitAll();
    __syncthreads();
    int wave = tid >> 5, lane = tid & 31;
    int laneLo = lane & 15, laneHi = lane >> 4;
    int wr = wave << 4;
    v8f acc2[8] = {};
    for (int c = 0; c < 4; ++c) {
        // phase 1: 128 cols of gelu(A@W1+b1) into LDS
#pragma unroll
        for (int nt = 0; nt < 8; ++nt) {
            v8f acc = {};
            int gcol = c * 128 + nt * 16 + laneLo;
#pragma unroll
            for (int kt = 0; kt < 4; ++kt) {
                int kb = kt << 5;
                const _Float16* pa = &ldsA[(wr + laneLo) * LDAH + kb + laneHi * 8];
                FragU fa; fa.u[0] = *(const v4u*)pa; fa.u[1] = *(const v4u*)(pa + 16);
                const _Float16* pb = &W1t[(size_t)gcol * DIM + kb + laneHi * 8];
                FragU fb; fb.u[0] = *(const v4u*)pb; fb.u[1] = *(const v4u*)(pb + 16);
                acc = __builtin_amdgcn_wmma_f32_16x16x32_f16(false, fa.h, false, fb.h,
                                                             (short)0, acc, false, false);
            }
            float bb = b1[gcol];
#pragma unroll
            for (int j = 0; j < 8; ++j) {
                int r = wr + laneHi * 8 + j;
                float u = acc[j] + bb;
                float gl = 0.5f * u * (1.0f + erff(u * 0.70710678f));   // exact gelu
                ldsH[r * LDAH + nt * 16 + laneLo] = (_Float16)gl;
            }
        }
        __syncthreads();
        // phase 2: partial ff2 accumulate over this K-chunk
#pragma unroll
        for (int nt = 0; nt < 8; ++nt) {
            v8f acc = acc2[nt];
#pragma unroll
            for (int kt = 0; kt < 4; ++kt) {
                int kb = kt << 5;
                const _Float16* pa = &ldsH[(wr + laneLo) * LDAH + kb + laneHi * 8];
                FragU fa; fa.u[0] = *(const v4u*)pa; fa.u[1] = *(const v4u*)(pa + 16);
                const _Float16* pb = &W2t[(size_t)(nt * 16 + laneLo) * FFDIM + c * 128 + kb + laneHi * 8];
                FragU fb; fb.u[0] = *(const v4u*)pb; fb.u[1] = *(const v4u*)(pb + 16);
                acc = __builtin_amdgcn_wmma_f32_16x16x32_f16(false, fa.h, false, fb.h,
                                                             (short)0, acc, false, false);
            }
            acc2[nt] = acc;
        }
        __syncthreads();
    }
    // epilogue: residual add
#pragma unroll
    for (int nt = 0; nt < 8; ++nt) {
        int col = nt * 16 + laneLo;
        float bb = b2[col];
        v8f a = acc2[nt];
#pragma unroll
        for (int j = 0; j < 8; ++j) {
            size_t row = rowBase + wr + laneHi * 8 + j;
            h[row * DIM + col] += a[j] + bb;
        }
    }
}

// ---------------- host launcher ----------------
extern "C" void kernel_launch(void* const* d_in, const int* in_sizes, int n_in,
                              void* d_out, int out_size, void* d_ws, size_t ws_size,
                              hipStream_t stream) {
    auto F = [&](int i) { return (const float*)d_in[i]; };
    const float *x, *embW, *embB, *pos, *trajW, *trajB;
    const float *normPG, *normPB, *normG, *normB;
    struct Blk {
        const float *qtW, *ktW, *qsW, *ksW, *wsig, *vW, *outW, *outB;
        const float *n1pg, *n1pb, *n1g, *n1b, *n2pg, *n2pb, *n2g, *n2b;
        const float *ff1W, *ff1b, *ff2W, *ff2b;
    } blk[3];

    bool insertion = (n_in > 0 && in_sizes[0] == NTOK * 12);
    if (insertion) {
        // depth-first insertion-order flattening: x, embed, pos, blocks..., norm, traj
        x = F(0); embW = F(1); embB = F(2); pos = F(3);
        for (int l = 0; l < 3; ++l) {
            int s = 4 + 20 * l;
            blk[l].qtW = F(s + 0);  blk[l].ktW = F(s + 1);  blk[l].qsW = F(s + 2);
            blk[l].ksW = F(s + 3);  blk[l].wsig = F(s + 4); blk[l].vW = F(s + 5);
            blk[l].outW = F(s + 6); blk[l].outB = F(s + 7);
            blk[l].n1pg = F(s + 8); blk[l].n1pb = F(s + 9); blk[l].n1g = F(s + 10); blk[l].n1b = F(s + 11);
            blk[l].n2pg = F(s + 12); blk[l].n2pb = F(s + 13); blk[l].n2g = F(s + 14); blk[l].n2b = F(s + 15);
            blk[l].ff1W = F(s + 16); blk[l].ff1b = F(s + 17); blk[l].ff2W = F(s + 18); blk[l].ff2b = F(s + 19);
        }
        normPG = F(64); normPB = F(65); normG = F(66); normB = F(67);
        trajW = F(68); trajB = F(69);
    } else {
        // jax tree_leaves (sorted dict keys): params first (blocks sorted inside), x last
        for (int l = 0; l < 3; ++l) {
            int s = 20 * l;
            blk[l].ff1W = F(s + 0); blk[l].ff1b = F(s + 1); blk[l].ff2W = F(s + 2); blk[l].ff2b = F(s + 3);
            blk[l].ksW = F(s + 4);  blk[l].ktW = F(s + 5);
            blk[l].n1b = F(s + 6);  blk[l].n1g = F(s + 7);  blk[l].n1pb = F(s + 8);  blk[l].n1pg = F(s + 9);
            blk[l].n2b = F(s + 10); blk[l].n2g = F(s + 11); blk[l].n2pb = F(s + 12); blk[l].n2pg = F(s + 13);
            blk[l].outW = F(s + 14); blk[l].outB = F(s + 15);
            blk[l].qsW = F(s + 16); blk[l].qtW = F(s + 17); blk[l].vW = F(s + 18); blk[l].wsig = F(s + 19);
        }
        embW = F(60); embB = F(61);
        normB = F(62); normG = F(63); normPB = F(64); normPG = F(65);
        pos = F(66); trajW = F(67); trajB = F(68); x = F(69);
    }

    // ---- workspace carve (256B aligned) ----
    uint8_t* wbase = (uint8_t*)d_ws;
    size_t off = 0;
    auto carve = [&](size_t bytes) -> void* {
        void* p = wbase + off;
        off = (off + bytes + 255) & ~(size_t)255;
        return p;
    };
    float*    h    = (float*)carve((size_t)NTOK * DIM * 4);
    _Float16* ln   = (_Float16*)carve((size_t)NTOK * DIM * 2);   // also reused as ctx / lnlast
    _Float16* qkv  = (_Float16*)carve((size_t)NTOK * NQKV * 2);
    float*    part = (float*)carve((size_t)(NTOK / 8) * 4);
    float*    epsv = (float*)carve(256);
    _Float16 *wqkv[3], *wout[3], *wff1[3], *wff2[3];
    for (int l = 0; l < 3; ++l) {
        wqkv[l] = (_Float16*)carve((size_t)NQKV * DIM * 2);
        wout[l] = (_Float16*)carve((size_t)DIM * DIM * 2);
        wff1[l] = (_Float16*)carve((size_t)FFDIM * DIM * 2);
        wff2[l] = (_Float16*)carve((size_t)DIM * FFDIM * 2);
    }
    _Float16* wtraj = (_Float16*)carve((size_t)DIM * DIM * 2);
    (void)ws_size; (void)out_size;

    (void)hipFuncSetAttribute(reinterpret_cast<const void*>(ffn_kernel),
                              hipFuncAttributeMaxDynamicSharedMemorySize, FFN_LDS_BYTES);

    // ---- weight packing (f16, transposed [N][K]) ----
    auto packLaunch = [&](const float* W, _Float16* outp, int K, int N, int Nrows) {
        int n = Nrows * K;
        packWT<<<dim3((n + 255) / 256), dim3(256), 0, stream>>>(W, outp, K, N, Nrows);
    };
    for (int l = 0; l < 3; ++l) {
        packLaunch(blk[l].qtW, wqkv[l] + 0 * DIM,   DIM, 64, 64);    // Q heads 0-1 (time)
        packLaunch(blk[l].qsW, wqkv[l] + 64 * DIM,  DIM, 64, 64);    // Q heads 2-3 (space)
        packLaunch(blk[l].ktW, wqkv[l] + 128 * DIM, DIM, 64, 64);
        packLaunch(blk[l].ksW, wqkv[l] + 192 * DIM, DIM, 64, 64);
        packLaunch(blk[l].vW,  wqkv[l] + 256 * DIM, DIM, DIM, DIM);
        packLaunch(blk[l].outW, wout[l], DIM, DIM, DIM);
        packLaunch(blk[l].ff1W, wff1[l], DIM, FFDIM, FFDIM);
        packLaunch(blk[l].ff2W, wff2[l], FFDIM, DIM, DIM);
    }
    packLaunch(trajW, wtraj, DIM, TOUTDIM, DIM);   // pad cols 120..127 with zeros

    // ---- forward pass ----
    const int mlnGrid = NTOK / 8;     // 22528
    const int gemmGrid = NTOK / 128;  // 1408

    embed_kernel<<<dim3((NTOK * DIM + 255) / 256), 256, 0, stream>>>(x, embW, embB, pos, h);

    for (int l = 0; l < 3; ++l) {
        // minkowski LN #1
        mln_pass1<<<mlnGrid, 256, 0, stream>>>(h, blk[l].n1pg, blk[l].n1pb, part);
        reduce_eps<<<1, 256, 0, stream>>>(part, mlnGrid, epsv);
        mln_pass2<<<mlnGrid, 256, 0, stream>>>(h, blk[l].n1pg, blk[l].n1pb,
                                               blk[l].n1g, blk[l].n1b, epsv, ln, 0);
        // QKV projection (WMMA)
        gemm128<0><<<gemmGrid, 256, GEMM_LDS_BYTES, stream>>>(ln, wqkv[l], nullptr,
                                                              (void*)qkv, nullptr,
                                                              NQKV, NQKV, 0);
        // attention (ctx aliases ln; ln fully consumed by the QKV GEMM)
        attn_kernel<<<dim3(BATCH * 4 / 8), 256, 0, stream>>>(qkv, ln, blk[l].wsig);
        // out-proj + bias + residual (WMMA)
        gemm128<1><<<gemmGrid, 256, GEMM_LDS_BYTES, stream>>>(ln, wout[l], blk[l].outB,
                                                              nullptr, h, DIM, DIM, 0);
        // minkowski LN #2
        mln_pass1<<<mlnGrid, 256, 0, stream>>>(h, blk[l].n2pg, blk[l].n2pb, part);
        reduce_eps<<<1, 256, 0, stream>>>(part, mlnGrid, epsv);
        mln_pass2<<<mlnGrid, 256, 0, stream>>>(h, blk[l].n2pg, blk[l].n2pb,
                                               blk[l].n2g, blk[l].n2b, epsv, ln, 0);
        // fused FFN + residual (WMMA)
        ffn_kernel<<<gemmGrid, 256, FFN_LDS_BYTES, stream>>>(ln, wff1[l], blk[l].ff1b,
                                                             wff2[l], blk[l].ff2b, h);
    }

    // final norm (global eps over all tokens, output only last token per sequence)
    mln_pass1<<<mlnGrid, 256, 0, stream>>>(h, normPG, normPB, part);
    reduce_eps<<<1, 256, 0, stream>>>(part, mlnGrid, epsv);
    mln_pass2<<<mlnGrid, 256, 0, stream>>>(h, normPG, normPB, normG, normB, epsv, ln, 1);

    // trajectory head: [16384 x 128] @ [128 x 120] + b  (WMMA, padded N)
    gemm128<2><<<dim3(BATCH / 128), 256, GEMM_LDS_BYTES, stream>>>(ln, wtraj, trajB,
                                                                   d_out, nullptr,
                                                                   DIM, DIM, TOUTDIM);
}
